// Attention_aux_63728724738413
// MI455X (gfx1250) — matrix operand order repaired and checked by reading, not compiled, not run on hardware
//
#include <hip/hip_runtime.h>
#include <math.h>

typedef __attribute__((ext_vector_type(16))) __bf16 v16bf;
typedef __attribute__((ext_vector_type(8)))  float  v8f;

#define DIM   1024
#define ENCD  3072
#define FTOT  4096
#define BATCH 32
#define SLEN  1024
#define ROWS  (BATCH * SLEN)   // 32768

#define BM 128
#define BN 256
#define BK 32
#define NTHREADS 512           // 16 waves

__device__ __forceinline__ float tanh_fast(float x) {
    // tanh(x) = 1 - 2/(exp(2x)+1); exact saturation at +/-inf, branch-free
    return 1.f - 2.f / (__expf(2.f * x) + 1.f);
}

// global fp32 -> registers for one K-step tile
__device__ __forceinline__ void load_tiles(
    int k0, int grow, int gb, int acol, int brow, int bcol, int n0,
    const float* __restrict__ enc, const float* __restrict__ hid,
    const float* __restrict__ W1, float4 a[2], float4 b[4])
{
    const float* asrc = (k0 < ENCD)
        ? enc + (size_t)grow * ENCD + (size_t)(k0 + acol)
        : hid + (size_t)gb * DIM + (size_t)(k0 - ENCD + acol);
    a[0] = ((const float4*)asrc)[0];
    a[1] = ((const float4*)asrc)[1];
    const float* bsrc = W1 + (size_t)(n0 + brow) * FTOT + (size_t)(k0 + bcol);
    b[0] = ((const float4*)bsrc)[0];
    b[1] = ((const float4*)bsrc)[1];
    b[2] = ((const float4*)bsrc)[2];
    b[3] = ((const float4*)bsrc)[3];
}

// registers -> LDS, fp32 -> bf16 via hardware converts
__device__ __forceinline__ void store_tiles(
    __bf16 (*__restrict__ As)[BK], __bf16 (*__restrict__ Bs)[BK],
    int arow, int acol, int brow, int bcol, const float4 a[2], const float4 b[4])
{
#pragma unroll
    for (int q = 0; q < 2; ++q) {
        As[arow][acol + 4 * q + 0] = (__bf16)a[q].x;
        As[arow][acol + 4 * q + 1] = (__bf16)a[q].y;
        As[arow][acol + 4 * q + 2] = (__bf16)a[q].z;
        As[arow][acol + 4 * q + 3] = (__bf16)a[q].w;
    }
#pragma unroll
    for (int q = 0; q < 4; ++q) {
        Bs[brow][bcol + 4 * q + 0] = (__bf16)b[q].x;
        Bs[brow][bcol + 4 * q + 1] = (__bf16)b[q].y;
        Bs[brow][bcol + 4 * q + 2] = (__bf16)b[q].z;
        Bs[brow][bcol + 4 * q + 3] = (__bf16)b[q].w;
    }
}

// ---------------------------------------------------------------------------
// Fused score GEMM: e[bs] += w2 . tanh(inputs @ W1^T + b1)
//   inputs[bs, f] = f < 3072 ? enc[bs, f] : hidden[b, f-3072]
//   bf16 WMMA 16x16x32, f32 accum, LDS double-buffered, 1 barrier / K-step.
//   Block tile 128(M) x 256(N); wave tile 32(M) x 64(N) (2x4 fragments).
//   Grid: (DIM/BN, ROWS/BM), 512 threads.
// ---------------------------------------------------------------------------
extern "C" __global__ __launch_bounds__(NTHREADS)
void aux_score_gemm(const float* __restrict__ enc,
                    const float* __restrict__ hid,
                    const float* __restrict__ W1,
                    const float* __restrict__ b1,
                    const float* __restrict__ w2,
                    float* __restrict__ e)
{
    __shared__ __bf16 As[2][BM][BK];   // 16 KB
    __shared__ __bf16 Bs[2][BN][BK];   // 32 KB

    const int tid  = threadIdx.x;
    const int lane = tid & 31;
    const int wv   = tid >> 5;     // wave 0..15
    const int wm   = wv & 3;       // 4 M groups of 32 rows
    const int wn   = wv >> 2;      // 4 N groups of 64 cols

    const int n0 = blockIdx.x * BN;
    const int m0 = blockIdx.y * BM;

    // staging assignments (512 threads)
    const int arow = tid >> 2;           // 0..127
    const int acol = (tid & 3) * 8;      // 0,8,16,24
    const int brow = tid >> 1;           // 0..255
    const int bcol = (tid & 1) * 16;     // 0,16

    const int grow = m0 + arow;          // global (b*S + s) row for A staging
    const int gb   = grow >> 10;

    const int ml = lane & 15;
    const int hl = lane >> 4;

    v8f acc[2][4];
#pragma unroll
    for (int mi = 0; mi < 2; ++mi)
#pragma unroll
        for (int t = 0; t < 4; ++t)
            acc[mi][t] = (v8f){0, 0, 0, 0, 0, 0, 0, 0};

    // prologue: stage K-step 0 into buffer 0
    float4 ra[2], rb[4];
    load_tiles(0, grow, gb, acol, brow, bcol, n0, enc, hid, W1, ra, rb);
    store_tiles(As[0], Bs[0], arow, acol, brow, bcol, ra, rb);

    int buf = 0;
    for (int k0 = 0; k0 < FTOT; k0 += BK) {
        __syncthreads();                 // LDS[buf] ready for everyone

        const bool has_next = (k0 + BK) < FTOT;
        if (has_next)                    // issue next tile's global loads early
            load_tiles(k0 + BK, grow, gb, acol, brow, bcol, n0,
                       enc, hid, W1, ra, rb);

        // ---- load ALL fragments first (12x ds_load_b128, single wait),
        //      then run 8 back-to-back WMMAs.
        // A fragment (16x32 bf16): VGPR i, lane-half hl:
        //      K = (hl?16:0) + 2*i, +1
        union U { unsigned int u[8]; v16bf v; } fa[2], fb[4];
#pragma unroll
        for (int mi = 0; mi < 2; ++mi)
#pragma unroll
            for (int i = 0; i < 8; ++i) {
                int kk = (hl << 4) + (i << 1);
                fa[mi].u[i] =
                    *(const unsigned int*)&As[buf][wm * 32 + mi * 16 + ml][kk];
            }
        // B fragment (32x16 bf16): lane = N, VGPR i: K = (hl?16:0)+2i
#pragma unroll
        for (int t = 0; t < 4; ++t) {
            const unsigned int* bp =
                (const unsigned int*)&Bs[buf][wn * 64 + t * 16 + ml][hl << 4];
#pragma unroll
            for (int i = 0; i < 8; ++i) fb[t].u[i] = bp[i];
        }
#pragma unroll
        for (int t = 0; t < 4; ++t) {
            acc[0][t] = __builtin_amdgcn_wmma_f32_16x16x32_bf16(
                false, fa[0].v, false, fb[t].v, (short)0, acc[0][t], false, false);
            acc[1][t] = __builtin_amdgcn_wmma_f32_16x16x32_bf16(
                false, fa[1].v, false, fb[t].v, (short)0, acc[1][t], false, false);
        }

        if (has_next)                    // fill the other buffer (no hazard)
            store_tiles(As[buf ^ 1], Bs[buf ^ 1], arow, acol, brow, bcol, ra, rb);
        buf ^= 1;
    }

    // ---- epilogue: +b1, tanh, dot with w2, butterfly reduce, atomic into e
    float p[2][8];
#pragma unroll
    for (int mi = 0; mi < 2; ++mi)
#pragma unroll
        for (int r = 0; r < 8; ++r) p[mi][r] = 0.f;

#pragma unroll
    for (int t = 0; t < 4; ++t) {
        const int n = n0 + wn * 64 + t * 16 + ml;    // C layout: N = lane&15
        const float w2v  = w2[n];
        const float bias = b1[n];
#pragma unroll
        for (int mi = 0; mi < 2; ++mi)
#pragma unroll
            for (int r = 0; r < 8; ++r)              // M = hl*8 + r
                p[mi][r] += tanh_fast(acc[mi][t][r] + bias) * w2v;
    }
#pragma unroll
    for (int mask = 1; mask < 16; mask <<= 1)
#pragma unroll
        for (int mi = 0; mi < 2; ++mi)
#pragma unroll
            for (int r = 0; r < 8; ++r)
                p[mi][r] += __shfl_xor(p[mi][r], mask, 32);

    if (ml == 0) {
#pragma unroll
        for (int mi = 0; mi < 2; ++mi) {
            const int mrow = m0 + wm * 32 + mi * 16 + hl * 8;
#pragma unroll
            for (int r = 0; r < 8; ++r)
                atomicAdd(&e[mrow + r], p[mi][r]);
        }
    }
}

// ---------------------------------------------------------------------------
extern "C" __global__ __launch_bounds__(256)
void aux_zero(float* __restrict__ p, int n)
{
    int i = blockIdx.x * 256 + threadIdx.x;
    if (i < n) p[i] = 0.f;
}

// ---------------------------------------------------------------------------
// Softmax over S per batch row
// ---------------------------------------------------------------------------
extern "C" __global__ __launch_bounds__(256)
void aux_softmax(const float* __restrict__ e, float* __restrict__ alpha)
{
    __shared__ float red[256];
    const int b = blockIdx.x;
    const int tid = threadIdx.x;
    const float* row = e + b * SLEN;

    float m = -1e30f;
    for (int s = tid; s < SLEN; s += 256) m = fmaxf(m, row[s]);
    red[tid] = m; __syncthreads();
    for (int st = 128; st > 0; st >>= 1) {
        if (tid < st) red[tid] = fmaxf(red[tid], red[tid + st]);
        __syncthreads();
    }
    m = red[0]; __syncthreads();

    float sum = 0.f;
    for (int s = tid; s < SLEN; s += 256) sum += __expf(row[s] - m);
    red[tid] = sum; __syncthreads();
    for (int st = 128; st > 0; st >>= 1) {
        if (tid < st) red[tid] += red[tid + st];
        __syncthreads();
    }
    const float inv = 1.f / red[0];
    for (int s = tid; s < SLEN; s += 256)
        alpha[b * SLEN + s] = __expf(row[s] - m) * inv;
}

// ---------------------------------------------------------------------------
// context[b, f] = sum_s alpha[b, s] * enc[b, s, f]   (bandwidth pass, 402 MB)
// ---------------------------------------------------------------------------
extern "C" __global__ __launch_bounds__(256)
void aux_context(const float* __restrict__ enc, const float* __restrict__ alpha,
                 float* __restrict__ ctx)
{
    __shared__ float al[SLEN];
    const int b = blockIdx.y;
    const int f = blockIdx.x * 256 + threadIdx.x;
    for (int s = threadIdx.x; s < SLEN; s += 256) al[s] = alpha[b * SLEN + s];
    __syncthreads();

    const float* ep = enc + (size_t)b * SLEN * ENCD + f;
    float sum = 0.f;
    for (int s = 0; s < SLEN; ++s)
        sum = fmaf(al[s], ep[(size_t)s * ENCD], sum);
    ctx[b * ENCD + f] = sum;
}

// ---------------------------------------------------------------------------
// out[b, d] = ctx[b, :] . W3[d, :] + b3[d]   (tiny; W3 L2-resident)
// ---------------------------------------------------------------------------
extern "C" __global__ __launch_bounds__(256)
void aux_out(const float* __restrict__ ctx, const float* __restrict__ W3,
             const float* __restrict__ b3, float* __restrict__ out)
{
    __shared__ float c[ENCD];
    const int b = blockIdx.y;
    const int d = blockIdx.x * 256 + threadIdx.x;
    for (int f = threadIdx.x; f < ENCD; f += 256) c[f] = ctx[b * ENCD + f];
    __syncthreads();

    const float* wrow = W3 + (size_t)d * ENCD;
    float sum = b3[d];
    for (int f = 0; f < ENCD; ++f)
        sum = fmaf(c[f], wrow[f], sum);
    out[b * DIM + d] = sum;
}

// ---------------------------------------------------------------------------
extern "C" void kernel_launch(void* const* d_in, const int* in_sizes, int n_in,
                              void* d_out, int out_size, void* d_ws, size_t ws_size,
                              hipStream_t stream)
{
    const float* hid = (const float*)d_in[0];   // (32, 1024)
    const float* enc = (const float*)d_in[1];   // (32, 1024, 3072)
    const float* W1  = (const float*)d_in[2];   // (1024, 4096)
    const float* b1  = (const float*)d_in[3];   // (1024)
    const float* w2  = (const float*)d_in[4];   // (1, 1024)
    const float* W3  = (const float*)d_in[5];   // (1024, 3072)
    const float* b3  = (const float*)d_in[6];   // (1024)
    float* out = (float*)d_out;                 // (32, 1024)

    float* e     = (float*)d_ws;          // 32768 f32
    float* alpha = e + ROWS;              // 32768 f32
    float* ctx   = alpha + ROWS;          // 32*3072 f32
    (void)in_sizes; (void)n_in; (void)out_size; (void)ws_size;

    aux_zero<<<ROWS / 256, 256, 0, stream>>>(e, ROWS);
    aux_score_gemm<<<dim3(DIM / BN, ROWS / BM), NTHREADS, 0, stream>>>(
        enc, hid, W1, b1, w2, e);
    aux_softmax<<<BATCH, 256, 0, stream>>>(e, alpha);
    aux_context<<<dim3(ENCD / 256, BATCH), 256, 0, stream>>>(enc, alpha, ctx);
    aux_out<<<dim3(DIM / 256, BATCH), 256, 0, stream>>>(ctx, W3, b3, out);
}
